// GDAttention_37684043055673
// MI455X (gfx1250) — compile-verified, matrix operand order checked by reading, not run on hardware
//
#include <hip/hip_runtime.h>
#include <hip/hip_bf16.h>

typedef __attribute__((ext_vector_type(16))) __bf16 v16bf;
typedef __attribute__((ext_vector_type(8)))  __bf16 v8bf;
typedef __attribute__((ext_vector_type(8)))  float  v8f;
typedef __attribute__((ext_vector_type(4)))  unsigned int v4u;
typedef __attribute__((ext_vector_type(8)))  int    v8i_;
typedef __attribute__((ext_vector_type(4)))  int    v4i_;

#define B_   2
#define S_   1024
#define D_   1024
#define H_   8
#define HD_  8192
#define V_   32000

// Tensor Data Mover availability (device pass only; arity differs by toolchain).
#if defined(__AMDGCN__) && defined(__has_builtin)
#if __has_builtin(__builtin_amdgcn_tensor_load_to_lds) && __has_builtin(__builtin_amdgcn_s_wait_tensorcnt)
#define HAVE_TDM 1
#endif
#endif
#ifndef HAVE_TDM
#define HAVE_TDM 0
#endif

// Build a 16-element bf16 A/B fragment from two contiguous 8-element halves
// (K = kofs..kofs+7 and 16+kofs..16+kofs+7). 16-byte aligned loads.
__device__ __forceinline__ v16bf frag16(const __bf16* p0) {
    v8bf lo = *(const v8bf*)(p0);
    v8bf hi = *(const v8bf*)(p0 + 16);
    return __builtin_shufflevector(lo, hi,
            0, 1, 2, 3, 4, 5, 6, 7, 8, 9, 10, 11, 12, 13, 14, 15);
}

// ---------------------------------------------------------------------------
// Stage 1: E_wte = mean(wte_weight, axis=0). Deterministic two-stage reduce.
// ---------------------------------------------------------------------------
__global__ void k_wte_partial(const float* __restrict__ wte, float* __restrict__ part) {
    int d = blockIdx.x * 256 + threadIdx.x;
    int c = blockIdx.y;                          // 0..124
    const float* pp = wte + (size_t)c * 256 * D_ + d;
    float s = 0.f;
    #pragma unroll 8
    for (int r = 0; r < 256; ++r) s += pp[(size_t)r * D_];
    part[c * D_ + d] = s;
}

__global__ void k_wte_final(const float* __restrict__ part, float* __restrict__ ewte) {
    int d = blockIdx.x * 256 + threadIdx.x;
    float s = 0.f;
    for (int c = 0; c < 125; ++c) s += part[c * D_ + d];
    ewte[d] = s * (1.0f / (float)V_);
}

// ---------------------------------------------------------------------------
// Stage 2: bias[s][d] = (e[0,s,d] + e[1,s,d] - 2*E_wte[d]) / (s+1)
// ---------------------------------------------------------------------------
__global__ void k_bias(const float* __restrict__ e, const float* __restrict__ ewte,
                       float* __restrict__ bias) {
    int idx = blockIdx.x * 256 + threadIdx.x;
    int s = idx >> 10;
    int d = idx & (D_ - 1);
    float v = e[idx] + e[(size_t)S_ * D_ + idx] - 2.0f * ewte[d];
    bias[idx] = v / (float)(s + 1);
}

// ---------------------------------------------------------------------------
// Stage 3a: W_oT[n][k] = bf16(W_o[k][n])
// ---------------------------------------------------------------------------
__global__ void k_wot(const float* __restrict__ wo, __bf16* __restrict__ wot) {
    size_t idx = (size_t)blockIdx.x * 256 + threadIdx.x;
    int k = (int)(idx & (HD_ - 1));
    int n = (int)(idx >> 13);
    wot[idx] = (__bf16)wo[(size_t)k * D_ + n];
}

// ---------------------------------------------------------------------------
// Stage 3b: Qb[b,h,q,d] = bf16(p[b,1+q,d] * wq[h,d] * wk[h,d])   (fold wq*wk)
// ---------------------------------------------------------------------------
__global__ void k_prep_q(const float* __restrict__ p, const float* __restrict__ wq,
                         const float* __restrict__ wk, __bf16* __restrict__ qb) {
    int idx = blockIdx.x * 256 + threadIdx.x;    // over B*H*S*D = 2^24
    int d = idx & 1023;
    int q = (idx >> 10) & 1023;
    int h = (idx >> 20) & 7;
    int b = idx >> 23;
    float v = p[((size_t)b * (S_ + 1) + 1 + q) * D_ + d];
    qb[idx] = (__bf16)(v * wq[h * D_ + d] * wk[h * D_ + d]);
}

// ---------------------------------------------------------------------------
// Stage 3c: Kb[b,k,d] = bf16(p[b,k,d])   (head-independent)
// ---------------------------------------------------------------------------
__global__ void k_prep_k(const float* __restrict__ p, __bf16* __restrict__ kb) {
    int idx = blockIdx.x * 256 + threadIdx.x;    // over B*S*D = 2^21
    int d = idx & 1023;
    int k = (idx >> 10) & 1023;
    int b = idx >> 20;
    kb[idx] = (__bf16)p[((size_t)b * (S_ + 1) + k) * D_ + d];
}

// ---------------------------------------------------------------------------
// Stage 3d: Vt[b,h,d,k] = bf16((e[b,k,d]-E_wte[d]) * wv[h,d])  (d-major so the
// AV B-fragment is a contiguous per-lane load). 16x16 LDS tile transpose.
// ---------------------------------------------------------------------------
__global__ void k_prep_vt(const float* __restrict__ e, const float* __restrict__ ewte,
                          const float* __restrict__ wv, __bf16* __restrict__ vt) {
    __shared__ float tile[16][17];
    int tx = threadIdx.x & 15;
    int ty = threadIdx.x >> 4;
    int d0 = blockIdx.x * 16;
    int k0 = blockIdx.y * 16;
    int h  = blockIdx.z & 7;
    int b  = blockIdx.z >> 3;
    int d = d0 + tx;
    float v = e[((size_t)b * S_ + (k0 + ty)) * D_ + d];
    tile[ty][tx] = (v - ewte[d]) * wv[h * D_ + d];
    __syncthreads();
    vt[((size_t)(b * H_ + h) * D_ + (d0 + ty)) * S_ + k0 + tx] = (__bf16)tile[tx][ty];
}

// ---------------------------------------------------------------------------
// Stage 4: flash attention, bf16 WMMA, TDM-staged K tiles (double-buffered).
// Block = 128 threads (4 waves), one block per (b, h, 16-query tile).
// Dynamic LDS layout (bytes):
//   [0,      64K)  K tile buffer 0 (32 keys x 1024 d, bf16)
//   [64K,   128K)  K tile buffer 1
//   [128K,  136K)  score partials float[4][16][32]
//   [136K, +1K  )  P tile bf16[16][32]
//   then alpha[16], m[16], l[16] (float)
// ---------------------------------------------------------------------------
#define SM_KB0   0
#define SM_KB1   65536
#define SM_SRED  131072
#define SM_SPB   (131072 + 8192)
#define SM_SA    (SM_SPB + 1024)
#define SM_SM    (SM_SA + 64)
#define SM_SL    (SM_SM + 64)
#define SM_TOTAL (SM_SL + 64)

__global__ __launch_bounds__(128)
void k_attn(const __bf16* __restrict__ qb, const __bf16* __restrict__ kb,
            const __bf16* __restrict__ vt, const float* __restrict__ wlr,
            __bf16* __restrict__ yout) {
    extern __shared__ __align__(16) char smem[];
    // NOTE: no aggregates of LDS-derived pointers (ld.lld cannot relocate
    // addrspacecast constants in static initializers) — select buffers by
    // byte offset instead.
    float*  sred   = (float*)(smem + SM_SRED);     // [w][r][j] -> w*512+r*32+j
    __bf16* sPb    = (__bf16*)(smem + SM_SPB);     // [r][j] -> r*32+j
    float*  sAlpha = (float*)(smem + SM_SA);
    float*  sM     = (float*)(smem + SM_SM);
    float*  sL     = (float*)(smem + SM_SL);

    const int lane  = threadIdx.x & 31;
    const int wave  = threadIdx.x >> 5;
    const int qt    = blockIdx.x & 63;
    const int h     = (blockIdx.x >> 6) & 7;
    const int b     = blockIdx.x >> 9;
    const int q0    = qt << 4;
    const int dbase = wave << 8;
    const int nrow  = lane & 15;
    const int kofs  = (lane & 16) ? 8 : 0;
    const int mrow  = (lane & 16) ? 8 : 0;

    const __bf16* kbb = kb + (size_t)b * S_ * D_;

    // ---- K tile staging: TDM (async DMA) or cooperative-copy fallback ------
    auto stage_k = [&](int k0, int buf_off) {
#if HAVE_TDM
        if (wave == 0) {
            // Low 32 bits of the flat shared-aperture address are the LDS
            // byte offset (ISA 10.2).
            unsigned lds_addr = (unsigned)(size_t)(void*)(smem + buf_off);
            unsigned long long ga =
                (unsigned long long)(size_t)(const void*)(kbb + (size_t)k0 * D_);
            v4u g0;
            g0[0] = 1u;                                   // count=1, user mode
            g0[1] = lds_addr;                             // lds_addr
            g0[2] = (unsigned)(ga & 0xFFFFFFFFu);         // global_addr lo
            g0[3] = (unsigned)((ga >> 32) & 0x1FFFFFFu)   // global_addr hi
                  | (2u << 30);                           // type = 2 (image)
            v8i_ g1;
            g1[0] = 0x00010000;            // wg_mask=0, data_size=1 (2 bytes)
            g1[1] = (int)((unsigned)D_ << 16);   // tensor_dim0 = 1024 (lo16)
            g1[2] = (int)(32u << 16);            // dim0 hi=0 | tensor_dim1 = 32
            g1[3] = (int)((unsigned)D_ << 16);   // dim1 hi=0 | tile_dim0 = 1024
            g1[4] = 32;                          // tile_dim1 = 32, tile_dim2 = 0
            g1[5] = D_;                          // tensor_dim0_stride = 1024
            g1[6] = 0;
            g1[7] = 0;
            v4i_ zz; zz[0] = 0; zz[1] = 0; zz[2] = 0; zz[3] = 0;
#if __clang_major__ >= 23
            v8i_ z8; for (int i = 0; i < 8; ++i) z8[i] = 0;
            __builtin_amdgcn_tensor_load_to_lds(g0, g1, zz, zz, z8, 0);
#else
            __builtin_amdgcn_tensor_load_to_lds(g0, g1, zz, zz, 0);
#endif
        }
#else
        const uint4* src = (const uint4*)(kbb + (size_t)k0 * D_);
        uint4* dv = (uint4*)(smem + buf_off);
        #pragma unroll 4
        for (int i = threadIdx.x; i < (32 * D_) / 8; i += 128) dv[i] = src[i];
#endif
    };

    // ---- preload Q fragments (already scaled by wq*wk) ----------------------
    const __bf16* qrow = qb + ((size_t)(b * H_ + h) * S_ + q0 + nrow) * D_ + dbase;
    v16bf qf[8];
    #pragma unroll
    for (int f = 0; f < 8; ++f) qf[f] = frag16(qrow + f * 32 + kofs);

    v8f acc[16];
    #pragma unroll
    for (int c = 0; c < 16; ++c)
        #pragma unroll
        for (int i = 0; i < 8; ++i) acc[c][i] = 0.f;

    if (threadIdx.x < 16) { sM[threadIdx.x] = -1e30f; sL[threadIdx.x] = 0.f; }

    stage_k(0, SM_KB0);

    const int kend = q0 + 16;
    for (int k0 = 0; k0 < kend; k0 += 32) {
        const int cur_off = ((k0 >> 5) & 1) ? SM_KB1 : SM_KB0;
        const int nxt_off = ((k0 >> 5) & 1) ? SM_KB0 : SM_KB1;
#if HAVE_TDM
        if (wave == 0) __builtin_amdgcn_s_wait_tensorcnt(0);
#endif
        __syncthreads();                      // tile k0 ready; sred/sPb free
        if (k0 + 32 < kend) stage_k(k0 + 32, nxt_off);        // async prefetch

        // ---- partial scores over this wave's 256-wide D slice --------------
        const __bf16* kt = (const __bf16*)(smem + cur_off);
        v8f c0, c1;
        #pragma unroll
        for (int i = 0; i < 8; ++i) { c0[i] = 0.f; c1[i] = 0.f; }
        #pragma unroll
        for (int f = 0; f < 8; ++f) {
            v16bf b0 = frag16(kt + (size_t)nrow * D_        + dbase + f * 32 + kofs);
            v16bf b1 = frag16(kt + (size_t)(16 + nrow) * D_ + dbase + f * 32 + kofs);
            c0 = __builtin_amdgcn_wmma_f32_16x16x32_bf16(false, qf[f], false, b0,
                                                         (short)0, c0, false, false);
            c1 = __builtin_amdgcn_wmma_f32_16x16x32_bf16(false, qf[f], false, b1,
                                                         (short)0, c1, false, false);
        }
        #pragma unroll
        for (int i = 0; i < 8; ++i) {
            sred[wave * 512 + (i + mrow) * 32 + nrow]      = c0[i];
            sred[wave * 512 + (i + mrow) * 32 + 16 + nrow] = c1[i];
        }
        __syncthreads();

        // ---- online softmax (wave 0, one lane per query row) ----------------
        if (wave == 0 && lane < 16) {
            int r = lane;
            int q = q0 + r;
            float mOld = sM[r], lOld = sL[r];
            float mNew = mOld;
            float sc[32];
            #pragma unroll
            for (int j = 0; j < 32; ++j) {
                float s = sred[r * 32 + j] + sred[512 + r * 32 + j]
                        + sred[1024 + r * 32 + j] + sred[1536 + r * 32 + j];
                if (k0 + j > q) s = -1e30f;            // causal mask
                sc[j] = s;
                mNew = fmaxf(mNew, s);
            }
            float alpha = __expf(mOld - mNew);
            float lsum = 0.f;
            #pragma unroll
            for (int j = 0; j < 32; ++j) {
                float pv = __expf(sc[j] - mNew);
                sPb[r * 32 + j] = (__bf16)pv;
                lsum += pv;
            }
            sM[r] = mNew;
            sL[r] = lOld * alpha + lsum;
            sAlpha[r] = alpha;
        }
        __syncthreads();

        // ---- P * V over this wave's 256 output columns ----------------------
        v16bf ap = frag16(sPb + nrow * 32 + kofs);
        float al[8];
        #pragma unroll
        for (int i = 0; i < 8; ++i) al[i] = sAlpha[i + mrow];
        #pragma unroll
        for (int c = 0; c < 16; ++c)
            #pragma unroll
            for (int i = 0; i < 8; ++i) acc[c][i] *= al[i];

        // Software-pipelined: fetch fragment c+1 before consuming fragment c
        // so WMMAs issue with loads still in flight (loadcnt>0) instead of
        // draining to zero every step.
        const __bf16* vrow = vt + ((size_t)(b * H_ + h) * D_ + dbase + nrow) * S_;
        v16bf vf = frag16(vrow + k0 + kofs);
        #pragma unroll
        for (int c = 0; c < 16; ++c) {
            v16bf vcur = vf;
            if (c < 15) vf = frag16(vrow + (size_t)(c + 1) * 16 * S_ + k0 + kofs);
            acc[c] = __builtin_amdgcn_wmma_f32_16x16x32_bf16(false, ap, false, vcur,
                                                             (short)0, acc[c], false, false);
        }
    }
    __syncthreads();

    // ---- finalize: 1/l, W_N = 1/(q+1), W_LR; store y as bf16 ----------------
    float wlrh = wlr[h];
    float scl[8];
    #pragma unroll
    for (int i = 0; i < 8; ++i) {
        int r = i + mrow;
        scl[i] = wlrh / (sL[r] * (float)(q0 + r + 1));
    }
    __bf16* yb = yout + ((size_t)(b * H_ + h) * S_ + q0) * D_ + dbase;
    #pragma unroll
    for (int c = 0; c < 16; ++c)
        #pragma unroll
        for (int i = 0; i < 8; ++i)
            yb[(size_t)(i + mrow) * D_ + c * 16 + nrow] = (__bf16)(acc[c][i] * scl[i]);
}

// ---------------------------------------------------------------------------
// Stage 5: out = y_flat(2048x8192 bf16) @ W_o + bias.
// One 64x16 C tile per wave (4 M-subtiles share each B fragment): 10 b128
// loads per 4 WMMAs, with the B fragment for k0+32 prefetched one step ahead.
// ---------------------------------------------------------------------------
__global__ __launch_bounds__(256)
void k_out(const __bf16* __restrict__ y, const __bf16* __restrict__ wot,
           const float* __restrict__ bias, float* __restrict__ out) {
    const int lane = threadIdx.x & 31;
    const int wave = threadIdx.x >> 5;
    const int tile = blockIdx.x * 8 + wave;    // 2048 wave-tiles
    const int tmq = tile & 31;                 // 32 row blocks of 64
    const int tn  = tile >> 5;                 // 64 col tiles of 16
    const int nrow = lane & 15;
    const int kofs = (lane & 16) ? 8 : 0;
    const int mrow = (lane & 16) ? 8 : 0;

    const __bf16* arow = y   + (size_t)(tmq * 64 + nrow) * HD_ + kofs;
    const __bf16* brow = wot + (size_t)(tn * 16 + nrow) * HD_ + kofs;

    v8f acc[4];
    #pragma unroll
    for (int j = 0; j < 4; ++j)
        #pragma unroll
        for (int i = 0; i < 8; ++i) acc[j][i] = 0.f;

    v16bf bm = frag16(brow);                   // pipelined B fragment
    for (int k0 = 0; k0 < HD_; k0 += 32) {
        v16bf aj[4];
        #pragma unroll
        for (int j = 0; j < 4; ++j)
            aj[j] = frag16(arow + (size_t)j * 16 * HD_ + k0);
        v16bf bcur = bm;
        if (k0 + 32 < HD_) bm = frag16(brow + k0 + 32);
        #pragma unroll
        for (int j = 0; j < 4; ++j)
            acc[j] = __builtin_amdgcn_wmma_f32_16x16x32_bf16(false, aj[j], false, bcur,
                                                             (short)0, acc[j], false, false);
        if (k0 + 256 < HD_) __builtin_prefetch(brow + k0 + 256, 0, 0);
    }

    const int col = tn * 16 + nrow;
    #pragma unroll
    for (int j = 0; j < 4; ++j) {
        const int row0 = tmq * 64 + j * 16 + mrow;
        #pragma unroll
        for (int i = 0; i < 8; ++i) {
            int rb = row0 + i;               // rb = b*S + s
            int s  = rb & (S_ - 1);
            out[(size_t)rb * D_ + col] = acc[j][i] + bias[(size_t)s * D_ + col];
        }
    }
}

// ---------------------------------------------------------------------------
// Workspace layout (1 MB = 1<<20 bytes):
//   [0, 4K)        E_wte           [4K, 516K)    wte partials
//   [1M, 5M)       bias f32        [8M, 40M)     y bf16 (B,H,S,D)
//   [40M, 56M)     W_oT bf16       [56M, 88M)    Qb bf16 (B,H,S,D)
//   [88M, 92M)     Kb bf16 (B,S,D) [92M, 124M)   Vt bf16 (B,H,D,S)
// ---------------------------------------------------------------------------
extern "C" void kernel_launch(void* const* d_in, const int* in_sizes, int n_in,
                              void* d_out, int out_size, void* d_ws, size_t ws_size,
                              hipStream_t stream) {
    const float* e   = (const float*)d_in[0];
    const float* p   = (const float*)d_in[1];
    const float* wte = (const float*)d_in[2];
    const float* wq  = (const float*)d_in[3];
    const float* wk  = (const float*)d_in[4];
    const float* wv  = (const float*)d_in[5];
    const float* wlr = (const float*)d_in[6];
    const float* wo  = (const float*)d_in[7];
    float* out = (float*)d_out;

    char* ws = (char*)d_ws;
    float*  ewte = (float*)(ws);
    float*  part = (float*)(ws + 4096);
    float*  bias = (float*)(ws + (1ull << 20));
    __bf16* yb   = (__bf16*)(ws + (8ull << 20));
    __bf16* wot  = (__bf16*)(ws + (40ull << 20));
    __bf16* qbuf = (__bf16*)(ws + (56ull << 20));
    __bf16* kbuf = (__bf16*)(ws + (88ull << 20));
    __bf16* vtb  = (__bf16*)(ws + (92ull << 20));

    k_wte_partial<<<dim3(4, 125), 256, 0, stream>>>(wte, part);
    k_wte_final<<<4, 256, 0, stream>>>(part, ewte);
    k_bias<<<(S_ * D_) / 256, 256, 0, stream>>>(e, ewte, bias);
    k_wot<<<(HD_ * D_) / 256, 256, 0, stream>>>(wo, wot);
    k_prep_q<<<(B_ * H_ * S_ * D_) / 256, 256, 0, stream>>>(p, wq, wk, qbuf);
    k_prep_k<<<(B_ * S_ * D_) / 256, 256, 0, stream>>>(p, kbuf);
    k_prep_vt<<<dim3(D_ / 16, S_ / 16, B_ * H_), 256, 0, stream>>>(e, ewte, wv, vtb);
    k_attn<<<B_ * H_ * (S_ / 16), 128, SM_TOTAL, stream>>>(qbuf, kbuf, vtb, wlr, yb);
    k_out<<<(2048) / 8, 256, 0, stream>>>(yb, wot, bias, out);
}
